// Attention_76673756168407
// MI455X (gfx1250) — compile-verified, matrix operand order checked by reading, not run on hardware
//
#include <hip/hip_runtime.h>

// ---------------------------------------------------------------------------
// Types for CDNA5 WMMA (wave32, 16x16x32 bf16 -> f32)
// ---------------------------------------------------------------------------
typedef __bf16 bf16_t;
typedef __attribute__((ext_vector_type(16))) __bf16 v16bf;
typedef __attribute__((ext_vector_type(8)))  __bf16 v8bf;
typedef __attribute__((ext_vector_type(8)))  float  v8f;

#define WMMA_BF16(a, b, c) \
  __builtin_amdgcn_wmma_f32_16x16x32_bf16(false, (a), false, (b), (short)0, (c), false, false)

static constexpr int Bb = 4, Ss = 2048, Dd = 1024, Hh = 16, HDm = 64;

// Load a 16(row) x 32(K) bf16 fragment from LDS in the CDNA5 A/B layout:
// lane l<16 : row=l,   K = 0..7 (v0..3) and 16..23 (v4..7)
// lane l>=16: row=l-16, K = 8..15 and 24..31
// (identical pattern for A tiles stored [M][K] and B tiles stored [N][K])
__device__ __forceinline__ v16bf load_frag16(const bf16_t* base, int stride) {
  const int lane = threadIdx.x & 31;
  const int lr = lane & 15;
  const int lh = lane >> 4;
  const bf16_t* p = base + lr * stride + lh * 8;
  v16bf f;
#pragma unroll
  for (int e = 0; e < 8; ++e) f[e] = p[e];
#pragma unroll
  for (int e = 0; e < 8; ++e) f[8 + e] = p[16 + e];
  return f;
}

// ---------------------------------------------------------------------------
// GEMM + bias: C[M,N] = A[M,K] @ W[K,N] + bias[N]   (bf16 WMMA, f32 accum).
// A input: f32 (converted while staging) or bf16 (raw b128 global->LDS copy).
// Output: f32 or bf16 (template).  Block tile 256x128, BK=32, 8 waves (4x2),
// wave tile 64x64 = 4x4 WMMA frags.  Double-buffered LDS, one barrier/K-step.
// Requires M%256==0, N%128==0, K%32==0 (true here).
// ---------------------------------------------------------------------------
struct StageF32  { float4 a[8]; float4 b[4]; };   // f32 A tile 256x32
struct StageBF16 { v8bf   a[4]; float4 b[4]; };   // bf16 A tile 256x32

template <typename T> struct StageFor;
template <> struct StageFor<float>  { using type = StageF32;  };
template <> struct StageFor<__bf16> { using type = StageBF16; };

__device__ __forceinline__ void gemm_gload_w(const float* __restrict__ W, int N,
                                             int n0, int k0, int t, float4* b) {
#pragma unroll
  for (int it = 0; it < 4; ++it) {              // W tile 32x128 = 1024 float4
    int idx = t + 256 * it;
    int kk  = idx >> 5;
    int n4  = (idx & 31) * 4;
    b[it] = *(const float4*)(W + (size_t)(k0 + kk) * N + n0 + n4);
  }
}

__device__ __forceinline__ void gemm_gload(const float* __restrict__ A,
                                           const float* __restrict__ W,
                                           int N, int K, int m0, int n0, int k0,
                                           int t, StageF32& s) {
#pragma unroll
  for (int it = 0; it < 8; ++it) {              // A tile 256x32 f32 = 2048 float4
    int idx = t + 256 * it;
    int r   = idx >> 3;
    int c4  = (idx & 7) * 4;
    s.a[it] = *(const float4*)(A + (size_t)(m0 + r) * K + k0 + c4);
  }
  gemm_gload_w(W, N, n0, k0, t, s.b);
}

__device__ __forceinline__ void gemm_gload(const bf16_t* __restrict__ A,
                                           const float* __restrict__ W,
                                           int N, int K, int m0, int n0, int k0,
                                           int t, StageBF16& s) {
#pragma unroll
  for (int it = 0; it < 4; ++it) {              // A tile 256x32 bf16 = 1024 x 16B
    int idx = t + 256 * it;
    int r   = idx >> 2;
    int c8  = (idx & 3) * 8;
    s.a[it] = *(const v8bf*)(A + (size_t)(m0 + r) * K + k0 + c8);
  }
  gemm_gload_w(W, N, n0, k0, t, s.b);
}

__device__ __forceinline__ void gemm_sstore_w(bf16_t (*Ws)[40], int t, const float4* b) {
#pragma unroll
  for (int it = 0; it < 4; ++it) {              // store W transposed: Ws[n][k]
    int idx = t + 256 * it;
    int kk  = idx >> 5;
    int n4  = (idx & 31) * 4;
    Ws[n4 + 0][kk] = (bf16_t)b[it].x; Ws[n4 + 1][kk] = (bf16_t)b[it].y;
    Ws[n4 + 2][kk] = (bf16_t)b[it].z; Ws[n4 + 3][kk] = (bf16_t)b[it].w;
  }
}

__device__ __forceinline__ void gemm_sstore(bf16_t (*As)[40], bf16_t (*Ws)[40],
                                            int t, const StageF32& s) {
#pragma unroll
  for (int it = 0; it < 8; ++it) {
    int idx = t + 256 * it;
    int r   = idx >> 3;
    int c4  = (idx & 7) * 4;
    As[r][c4 + 0] = (bf16_t)s.a[it].x; As[r][c4 + 1] = (bf16_t)s.a[it].y;
    As[r][c4 + 2] = (bf16_t)s.a[it].z; As[r][c4 + 3] = (bf16_t)s.a[it].w;
  }
  gemm_sstore_w(Ws, t, s.b);
}

__device__ __forceinline__ void gemm_sstore(bf16_t (*As)[40], bf16_t (*Ws)[40],
                                            int t, const StageBF16& s) {
#pragma unroll
  for (int it = 0; it < 4; ++it) {              // raw 16B copies, no conversion
    int idx = t + 256 * it;
    int r   = idx >> 2;
    int c8  = (idx & 3) * 8;
    *(v8bf*)&As[r][c8] = s.a[it];
  }
  gemm_sstore_w(Ws, t, s.b);
}

template <typename AT, typename OT>
__global__ __launch_bounds__(256)
void gemm_bias_kernel(const AT* __restrict__ A, const float* __restrict__ W,
                      const float* __restrict__ bias, OT* __restrict__ C,
                      int M, int N, int K) {
  __shared__ bf16_t As[2][256][40];   // [buf][m][k], padded (conflict-free b128)
  __shared__ bf16_t Ws[2][128][40];   // [buf][n][k], W transposed

  const int t    = threadIdx.x;
  const int lane = t & 31;
  const int wid  = t >> 5;
  const int wm   = (wid & 3) * 64;    // 4 waves over M
  const int wn   = (wid >> 2) * 64;   // 2 waves over N
  const int n0   = blockIdx.x * 128;
  const int m0   = blockIdx.y * 256;

  v8f acc[4][4] = {};
  typename StageFor<AT>::type st;

  gemm_gload(A, W, N, K, m0, n0, 0, t, st);
  gemm_sstore(As[0], Ws[0], t, st);
  int buf = 0;

  for (int k0 = 0; k0 < K; k0 += 32) {
    const bool nxt = (k0 + 32) < K;
    if (nxt) gemm_gload(A, W, N, K, m0, n0, k0 + 32, t, st);
    __syncthreads();                         // single barrier per K-step
    if (nxt) gemm_sstore(As[buf ^ 1], Ws[buf ^ 1], t, st);

    v16bf af[4], bfr[4];
#pragma unroll
    for (int i = 0; i < 4; ++i) af[i]  = load_frag16(&As[buf][wm + i * 16][0], 40);
#pragma unroll
    for (int i = 0; i < 4; ++i) bfr[i] = load_frag16(&Ws[buf][wn + i * 16][0], 40);
#pragma unroll
    for (int mi = 0; mi < 4; ++mi)
#pragma unroll
      for (int ni = 0; ni < 4; ++ni)
        acc[mi][ni] = WMMA_BF16(af[mi], bfr[ni], acc[mi][ni]);
    buf ^= 1;
  }

  // Epilogue: C/D layout -> lane = column (n), VGPR e -> row m = lh*8 + e
  const int lr = lane & 15, lh = lane >> 4;
#pragma unroll
  for (int ni = 0; ni < 4; ++ni) {
    int col  = n0 + wn + ni * 16 + lr;
    float bv = bias[col];
#pragma unroll
    for (int mi = 0; mi < 4; ++mi) {
      int row = m0 + wm + mi * 16 + lh * 8;
      OT* cp = C + (size_t)row * N + col;
#pragma unroll
      for (int e = 0; e < 8; ++e) cp[(size_t)e * N] = (OT)(acc[mi][ni][e] + bv);
    }
  }
}

// ---------------------------------------------------------------------------
// Flash attention (bf16 in / bf16 out), computed transposed so softmax stats
// are per-lane.  qkv layout: [B, S, 3*D] bf16, Q at +0, K at +D, V at +2D;
// head h uses columns h*64..h*64+63.  Block = 128 query rows of one (b,h);
// 8 waves, each owns 16 query rows (lane dim of all fragments).  Key block
// 64, double-buffered K/V staging -> one barrier per key block, 16 WMMAs.
//   S^T tile = K(64x64) @ Q^T           -> lane = q, VGPR = key
//   P^T (f32 D-frags) -> bf16 B-frags is PURELY per-lane on CDNA5
//   O^T += V^T(64x64) @ P^T(64x16)      -> lane = q, VGPR = d
// ---------------------------------------------------------------------------
struct KVStage { v8bf k[2]; v8bf v[2]; };

__device__ __forceinline__ void kv_gload(const bf16_t* __restrict__ qkv,
                                         int b, int h, int kb, int t, KVStage& s) {
#pragma unroll
  for (int it = 0; it < 2; ++it) {            // 64x64 bf16 = 512 x 16B each
    int idx = t + 256 * it;
    int j   = idx >> 3;
    int c8  = (idx & 7) * 8;
    size_t base = (size_t)(b * Ss + kb + j) * (3 * Dd) + h * HDm + c8;
    s.k[it] = *(const v8bf*)(qkv + base + Dd);
    s.v[it] = *(const v8bf*)(qkv + base + 2 * Dd);
  }
}

__device__ __forceinline__ void kv_sstore(bf16_t (*Ksb)[72], bf16_t (*Vtb)[72],
                                          int t, const KVStage& s) {
#pragma unroll
  for (int it = 0; it < 2; ++it) {
    int idx = t + 256 * it;
    int j   = idx >> 3;
    int c8  = (idx & 7) * 8;
    *(v8bf*)&Ksb[j][c8] = s.k[it];            // raw 16B copy
#pragma unroll
    for (int i = 0; i < 8; ++i) Vtb[c8 + i][j] = s.v[it][i];  // transpose scatter
  }
}

__global__ __launch_bounds__(256)
void flash_attn_kernel(const bf16_t* __restrict__ qkv, bf16_t* __restrict__ out) {
  __shared__ bf16_t Qs[128][72];       // [q][d]
  __shared__ bf16_t Ks[2][64][72];     // [buf][key][d]
  __shared__ bf16_t Vt[2][64][72];     // [buf][d][key]  (V transposed)

  const int t    = threadIdx.x;
  const int lane = t & 31;
  const int wq   = t >> 5;             // wave -> its 16 query rows
  const int b    = blockIdx.y >> 4;
  const int h    = blockIdx.y & 15;
  const int q0   = blockIdx.x * 128;

  // Stage Q tile 128x64 bf16 (1024 x 16B, 256 threads x 4) -- raw copies
#pragma unroll
  for (int it = 0; it < 4; ++it) {
    int idx = t + 256 * it;
    int r   = idx >> 3;
    int c8  = (idx & 7) * 8;
    *(v8bf*)&Qs[r][c8] =
        *(const v8bf*)(qkv + (size_t)(b * Ss + q0 + r) * (3 * Dd) + h * HDm + c8);
  }
  KVStage st;
  kv_gload(qkv, b, h, 0, t, st);
  __syncthreads();
  kv_sstore(Ks[0], Vt[0], t, st);
  // Q^T B-fragments for this wave's 16 rows (lane = q row), d halves
  const v16bf bq0 = load_frag16(&Qs[wq * 16][0], 72);
  const v16bf bq1 = load_frag16(&Qs[wq * 16][32], 72);

  float m_i = -INFINITY, l_i = 0.0f;
  v8f o[4] = {};                       // O^T accum: d rows [r*16 .. r*16+15]
  int buf = 0;

  for (int kb = 0; kb < Ss; kb += 64) {
    const bool nxt = (kb + 64) < Ss;
    if (nxt) kv_gload(qkv, b, h, kb + 64, t, st);
    __syncthreads();                   // single barrier per key block
    if (nxt) kv_sstore(Ks[buf ^ 1], Vt[buf ^ 1], t, st);

    // S^T = K @ Q^T : 4 key sub-blocks x (2 d-halves) = 8 WMMAs
    v8f sc[4];
#pragma unroll
    for (int kj = 0; kj < 4; ++kj) {
      v16bf aka = load_frag16(&Ks[buf][kj * 16][0], 72);
      v16bf akb = load_frag16(&Ks[buf][kj * 16][32], 72);
      v8f s0 = {};
      s0 = WMMA_BF16(aka, bq0, s0);
      s0 = WMMA_BF16(akb, bq1, s0);
      sc[kj] = s0;
    }

    // Online softmax over 32 keys/lane (partner lane ^16 holds the other 32)
    float tmax = -INFINITY;
#pragma unroll
    for (int kj = 0; kj < 4; ++kj)
#pragma unroll
      for (int e = 0; e < 8; ++e) {
        sc[kj][e] *= 0.125f;           // 1/sqrt(64)
        tmax = fmaxf(tmax, sc[kj][e]);
      }
    tmax = fmaxf(tmax, __shfl_xor(tmax, 16, 32));
    float m_new = fmaxf(m_i, tmax);
    float corr  = __expf(m_i - m_new);
    float rsum  = 0.0f;
    v16bf bp0, bp1;                    // P^T B-frags: keys 0..31 and 32..63
#pragma unroll
    for (int e = 0; e < 8; ++e) {
      float p0 = __expf(sc[0][e] - m_new);
      float p1 = __expf(sc[1][e] - m_new);
      float p2 = __expf(sc[2][e] - m_new);
      float p3 = __expf(sc[3][e] - m_new);
      rsum += (p0 + p1) + (p2 + p3);
      bp0[e] = (bf16_t)p0; bp0[8 + e] = (bf16_t)p1;
      bp1[e] = (bf16_t)p2; bp1[8 + e] = (bf16_t)p3;
    }
    rsum += __shfl_xor(rsum, 16, 32);
    l_i = l_i * corr + rsum;
    m_i = m_new;
#pragma unroll
    for (int r = 0; r < 4; ++r)
#pragma unroll
      for (int e = 0; e < 8; ++e) o[r][e] *= corr;

    // O^T += V^T @ P^T : 4 d sub-blocks x 2 key-halves = 8 WMMAs
#pragma unroll
    for (int r = 0; r < 4; ++r) {
      v16bf av0 = load_frag16(&Vt[buf][r * 16][0], 72);
      v16bf av1 = load_frag16(&Vt[buf][r * 16][32], 72);
      o[r] = WMMA_BF16(av0, bp0, o[r]);
      o[r] = WMMA_BF16(av1, bp1, o[r]);
    }
    buf ^= 1;
  }

  // Normalize and store: lane = q row, VGPR e -> d = r*16 + lh*8 + e
  float inv = 1.0f / l_i;
  const int lr = lane & 15, lh = lane >> 4;
  bf16_t* ob = out + (size_t)(b * Ss + q0 + wq * 16 + lr) * Dd + h * HDm + lh * 8;
#pragma unroll
  for (int r = 0; r < 4; ++r)
#pragma unroll
    for (int e = 0; e < 8; ++e) ob[r * 16 + e] = (bf16_t)(o[r][e] * inv);
}

// ---------------------------------------------------------------------------
// Host launch: QKV GEMM (f32->bf16) -> flash attention (bf16) -> projection
// (bf16->f32).  d_ws: [0,48MB) qkv bf16 [B,S,3D]; [48MB,64MB) attn bf16.
// ---------------------------------------------------------------------------
extern "C" void kernel_launch(void* const* d_in, const int* in_sizes, int n_in,
                              void* d_out, int out_size, void* d_ws, size_t ws_size,
                              hipStream_t stream) {
  (void)in_sizes; (void)n_in; (void)out_size; (void)ws_size;
  const float* x      = (const float*)d_in[0];
  const float* w_qkv  = (const float*)d_in[1];
  const float* b_qkv  = (const float*)d_in[2];
  const float* w_proj = (const float*)d_in[3];
  const float* b_proj = (const float*)d_in[4];
  float* outp = (float*)d_out;

  bf16_t* qkv  = (bf16_t*)d_ws;                     // B*S*3D bf16 (48MB)
  bf16_t* attn = qkv + (size_t)Bb * Ss * 3 * Dd;    // B*S*D  bf16 (16MB)

  const int M = Bb * Ss;   // 8192

  // 1) Fused QKV projection: [8192,1024]f32 @ [1024,3072] + bias -> bf16
  gemm_bias_kernel<float, bf16_t>
      <<<dim3((3 * Dd) / 128, M / 256), dim3(256), 0, stream>>>(
          x, w_qkv, b_qkv, qkv, M, 3 * Dd, Dd);

  // 2) Multi-head flash attention (bf16 -> bf16)
  flash_attn_kernel<<<dim3(Ss / 128, Bb * Hh), dim3(256), 0, stream>>>(qkv, attn);

  // 3) Output projection: [8192,1024]bf16 @ [1024,1024] + bias -> f32
  gemm_bias_kernel<bf16_t, float>
      <<<dim3(Dd / 128, M / 256), dim3(256), 0, stream>>>(
          attn, w_proj, b_proj, outp, M, Dd, Dd);
}
